// Net_27530740367671
// MI455X (gfx1250) — compile-verified, matrix operand order checked by reading, not run on hardware
//
#include <hip/hip_runtime.h>
#include <hip/hip_bf16.h>
#include <math.h>

#define BB   16
#define NP   4096
#define KNN  20
#define NCLS 40
#define NPTS (BB * NP)

typedef __attribute__((ext_vector_type(8)))  _Float16 v8h;
typedef __attribute__((ext_vector_type(16))) _Float16 v16h;
typedef __attribute__((ext_vector_type(8)))  float    v8f;

// ---------------------------------------------------------------- WMMA helpers

static __device__ __forceinline__ v8f wmma16(v16h a, v16h b, v8f c) {
  // D = A(16x32 f16) * B(32x16 f16) + C(16x16 f32)
  return __builtin_amdgcn_wmma_f32_16x16x32_f16(false, a, false, b, (short)0, c,
                                                false, false);
}

// Load a 16(row)x32(K) f16 fragment in the CDNA5 16-bit A/B VGPR layout from a
// row-major buffer (stride in halves). Lanes 0-15: K[kBase+0..7],K[kBase+16..23];
// lanes 16-31: K[kBase+8..15],K[kBase+24..31].
static __device__ __forceinline__ v16h load_frag(const _Float16* base, int stride,
                                                 int kBase, int lane) {
  int row  = lane & 15;
  int kOff = kBase + ((lane & 16) ? 8 : 0);
  const _Float16* r = base + row * stride + kOff;
  v8h lo = *(const v8h*)(r);
  v8h hi = *(const v8h*)(r + 16);
  v16h out;
#pragma unroll
  for (int i = 0; i < 8; ++i) { out[i] = lo[i]; out[i + 8] = hi[i]; }
  return out;
}

// Store one 16x16 f32 C tile (bias + optional relu) as f16 row-major.
static __device__ __forceinline__ void store_tile_f16(_Float16* dst, int stride,
                                                      v8f c, float bias, bool relu,
                                                      int lane) {
  int col   = lane & 15;
  int rbase = (lane & 16) ? 8 : 0;
#pragma unroll
  for (int i = 0; i < 8; ++i) {
    float v = c[i] + bias;
    if (relu) v = fmaxf(v, 0.0f);
    dst[(rbase + i) * stride + col] = (_Float16)v;
  }
}

// Column-wise max of a 16x16 C tile; valid in all lanes, col = lane & 15.
static __device__ __forceinline__ float tile_colmax(v8f c) {
  float m = c[0];
#pragma unroll
  for (int i = 1; i < 8; ++i) m = fmaxf(m, c[i]);
  m = fmaxf(m, __shfl_xor(m, 16, 32));
  return m;
}

// Device-scope float atomic max (non-returning) on global memory.
static __device__ __forceinline__ void gmaxf(float* addr, float v) {
  asm volatile("global_atomic_max_num_f32 %0, %1, off scope:SCOPE_DEV"
               :: "v"(addr), "v"(v) : "memory");
}

// Async memory->LDS 128-bit copy (ASYNCcnt-tracked DMA, no VGPR data path).
// lds: generic pointer to __shared__ (low 32 bits are the LDS offset),
// gaddr: 64-bit global address. GV addressing form.
static __device__ __forceinline__ void async_ld_b128(void* lds, const void* gaddr) {
  unsigned int l = (unsigned int)(uintptr_t)lds;
  asm volatile("global_load_async_to_lds_b128 %0, %1, off"
               :: "v"(l), "v"(gaddr) : "memory");
}
static __device__ __forceinline__ void async_wait0() {
  asm volatile("s_wait_asynccnt 0x0" ::: "memory");
}

// ---------------------------------------------------------------- prep kernels

__global__ __launch_bounds__(256) void k_prep_points(const float* __restrict__ pos,
                                                     const float* __restrict__ feat,
                                                     float* __restrict__ x0,
                                                     float* __restrict__ sq) {
  int t = blockIdx.x * 256 + threadIdx.x;
  if (t >= NPTS) return;
  float a = pos[t * 3 + 0], b = pos[t * 3 + 1], c = pos[t * 3 + 2], d = feat[t];
  x0[t * 4 + 0] = a; x0[t * 4 + 1] = b; x0[t * 4 + 2] = c; x0[t * 4 + 3] = d;
  sq[t] = a * a + b * b + c * c + d * d;
}

__global__ __launch_bounds__(256) void k_prep_weights(
    const float* W1, const float* b1, const float* g1, const float* be1,
    const float* m1, const float* v1,
    const float* W2, const float* b2, const float* g2, const float* be2,
    const float* m2, const float* v2,
    const float* W3, const float* Wc, const float* Wl,
    _Float16* W1t, _Float16* W2t, _Float16* W3t, _Float16* Wct, _Float16* Wlt,
    float* b1f, float* b2f, float* out2) {
  __shared__ float s1[64], s2[64];
  int t = threadIdx.x;
  if (t < 64) {
    float s = g1[t] * rsqrtf(v1[t] + 1e-5f);
    s1[t] = s; b1f[t] = (b1[t] - m1[t]) * s + be1[t];
    float q = g2[t] * rsqrtf(v2[t] + 1e-5f);
    s2[t] = q; b2f[t] = (b2[t] - m2[t]) * q + be2[t];
  }
  __syncthreads();
  // Transposed [N][K] layouts; BN scale folded into W1/W2 columns. W1 K 8->32 pad.
  for (int i = t; i < 64 * 32; i += 256) {
    int n = i / 32, k = i % 32;
    W1t[i] = (_Float16)((k < 8) ? W1[k * 64 + n] * s1[n] : 0.0f);
  }
  for (int i = t; i < 64 * 64; i += 256) {
    int n = i / 64, k = i % 64;
    W2t[i] = (_Float16)(W2[k * 64 + n] * s2[n]);
  }
  for (int i = t; i < 64 * 64; i += 256) {
    int n = i / 64, k = i % 64;
    W3t[i] = (_Float16)W3[k * 64 + n];
  }
  for (int i = t; i < 128 * 128; i += 256) {
    int n = i / 128, k = i % 128;
    Wct[i] = (_Float16)Wc[k * 128 + n];
  }
  for (int i = t; i < 1024 * 192; i += 256) {
    int n = i / 192, k = i % 192;
    Wlt[i] = (_Float16)Wl[k * 1024 + n];
  }
  for (int i = t; i < BB * 1024; i += 256) out2[i] = -3.0e38f;
}

// ---------------------------------------------------------------- kNN (top-20)

__global__ __launch_bounds__(256) void k_knn(const float* __restrict__ x0,
                                             const float* __restrict__ sq,
                                             int* __restrict__ idxb) {
  int q    = blockIdx.x * 256 + threadIdx.x;  // global query point
  int b    = q / NP;
  int base = b * NP;
  float4 xi = *(const float4*)(x0 + (size_t)q * 4);
  float sqi = sq[q];

  __shared__ float cx[128 * 4];
  __shared__ float cs[128];

  float bd[KNN]; int bi[KNN];
#pragma unroll
  for (int j = 0; j < KNN; ++j) { bd[j] = 3.0e38f; bi[j] = base; }
  float worst = 3.0e38f; int wslot = 0;

  for (int tile = 0; tile < NP; tile += 128) {
    __syncthreads();
    for (int i = threadIdx.x; i < 128 * 4; i += 256)
      cx[i] = x0[(size_t)(base + tile) * 4 + i];
    for (int i = threadIdx.x; i < 128; i += 256)
      cs[i] = sq[base + tile + i];
    __syncthreads();
    for (int j = 0; j < 128; ++j) {
      float dot = xi.x * cx[j * 4 + 0] + xi.y * cx[j * 4 + 1] +
                  xi.z * cx[j * 4 + 2] + xi.w * cx[j * 4 + 3];
      float d = sqi + cs[j] - 2.0f * dot;
      if (d < worst) {
        int cand = base + tile + j;
#pragma unroll
        for (int u = 0; u < KNN; ++u)
          if (u == wslot) { bd[u] = d; bi[u] = cand; }
        worst = bd[0]; wslot = 0;
#pragma unroll
        for (int u = 1; u < KNN; ++u)
          if (bd[u] > worst) { worst = bd[u]; wslot = u; }
      }
    }
  }
  for (int j = 0; j < KNN; ++j) idxb[(size_t)q * KNN + j] = bi[j];
}

// ---------------------------------------------------------------- EdgeConv 1

__global__ __launch_bounds__(256) void k_conv1(
    const float* __restrict__ x0, const int* __restrict__ idxb,
    const _Float16* __restrict__ W1t, const _Float16* __restrict__ W2t,
    const _Float16* __restrict__ W3t,
    const float* __restrict__ b1f, const float* __restrict__ b2f,
    const float* __restrict__ b3,
    float* __restrict__ x1, _Float16* __restrict__ xcat) {
  __shared__ alignas(16) _Float16 w1L[64 * 32];
  __shared__ alignas(16) _Float16 w2L[64 * 64];
  __shared__ alignas(16) _Float16 w3L[64 * 64];
  __shared__ alignas(16) _Float16 eL[8 * 32 * 32];
  __shared__ alignas(16) _Float16 hL[8 * 32 * 64];

  int tid = threadIdx.x, lane = tid & 31, wave = tid >> 5;
  // async DMA weight staging (ASYNCcnt tracked; overlaps with e-row build)
  for (int i = tid; i < 64 * 32 / 8; i += 256)
    async_ld_b128(w1L + i * 8, W1t + i * 8);
  for (int i = tid; i < 64 * 64 / 8; i += 256)
    async_ld_b128(w2L + i * 8, W2t + i * 8);
  for (int i = tid; i < 64 * 64 / 8; i += 256)
    async_ld_b128(w3L + i * 8, W3t + i * 8);

  int p = blockIdx.x * 8 + wave;  // one point per wave
  {  // build e rows: lane r handles edge row r (rows 20..31 duplicate edge 0)
    float4 xi = *(const float4*)(x0 + (size_t)p * 4);
    int r  = lane;
    int nb = idxb[(size_t)p * KNN + (r < KNN ? r : 0)];
    float4 xj = *(const float4*)(x0 + (size_t)nb * 4);
    _Float16* row = eL + wave * 32 * 32 + r * 32;
    row[0] = (_Float16)xi.x; row[1] = (_Float16)xi.y;
    row[2] = (_Float16)xi.z; row[3] = (_Float16)xi.w;
    row[4] = (_Float16)(xj.x - xi.x); row[5] = (_Float16)(xj.y - xi.y);
    row[6] = (_Float16)(xj.z - xi.z); row[7] = (_Float16)(xj.w - xi.w);
    v8h z = {};
    *(v8h*)(row + 8) = z; *(v8h*)(row + 16) = z; *(v8h*)(row + 24) = z;
  }
  async_wait0();
  __syncthreads();

  const _Float16* myE = eL + wave * 32 * 32;
  _Float16*       myH = hL + wave * 32 * 64;
  v8f zero = {};

  {  // layer 1: 32x8(pad32) @ 32x64
    v16h a0 = load_frag(myE, 32, 0, lane);
    v16h a1 = load_frag(myE + 16 * 32, 32, 0, lane);
#pragma unroll
    for (int nt = 0; nt < 4; ++nt) {
      v16h bf = load_frag(w1L + nt * 16 * 32, 32, 0, lane);
      v8f c0 = wmma16(a0, bf, zero);
      v8f c1 = wmma16(a1, bf, zero);
      float bias = b1f[nt * 16 + (lane & 15)];
      store_tile_f16(myH + 0 * 64 + nt * 16, 64, c0, bias, true, lane);
      store_tile_f16(myH + 16 * 64 + nt * 16, 64, c1, bias, true, lane);
    }
  }

  {  // layer 2: 32x64 @ 64x64
    v8f c2[2][4];
#pragma unroll
    for (int m = 0; m < 2; ++m)
#pragma unroll
      for (int nt = 0; nt < 4; ++nt) c2[m][nt] = zero;
#pragma unroll
    for (int ks = 0; ks < 2; ++ks) {
      v16h a0 = load_frag(myH, 64, ks * 32, lane);
      v16h a1 = load_frag(myH + 16 * 64, 64, ks * 32, lane);
#pragma unroll
      for (int nt = 0; nt < 4; ++nt) {
        v16h bf = load_frag(w2L + nt * 16 * 64, 64, ks * 32, lane);
        c2[0][nt] = wmma16(a0, bf, c2[0][nt]);
        c2[1][nt] = wmma16(a1, bf, c2[1][nt]);
      }
    }
#pragma unroll
    for (int nt = 0; nt < 4; ++nt) {
      float bias = b2f[nt * 16 + (lane & 15)];
      store_tile_f16(myH + 0 * 64 + nt * 16, 64, c2[0][nt], bias, true, lane);
      store_tile_f16(myH + 16 * 64 + nt * 16, 64, c2[1][nt], bias, true, lane);
    }
  }

  {  // layer 3: 32x64 @ 64x64, then max over edges, +b3
    v8f c3[2][4];
#pragma unroll
    for (int m = 0; m < 2; ++m)
#pragma unroll
      for (int nt = 0; nt < 4; ++nt) c3[m][nt] = zero;
#pragma unroll
    for (int ks = 0; ks < 2; ++ks) {
      v16h a0 = load_frag(myH, 64, ks * 32, lane);
      v16h a1 = load_frag(myH + 16 * 64, 64, ks * 32, lane);
#pragma unroll
      for (int nt = 0; nt < 4; ++nt) {
        v16h bf = load_frag(w3L + nt * 16 * 64, 64, ks * 32, lane);
        c3[0][nt] = wmma16(a0, bf, c3[0][nt]);
        c3[1][nt] = wmma16(a1, bf, c3[1][nt]);
      }
    }
#pragma unroll
    for (int nt = 0; nt < 4; ++nt) {
      float m = fmaxf(tile_colmax(c3[0][nt]), tile_colmax(c3[1][nt]));
      int col = nt * 16 + (lane & 15);
      m += b3[col];
      if (lane < 16) {
        x1[(size_t)p * 64 + col]    = m;
        xcat[(size_t)p * 192 + col] = (_Float16)m;
      }
    }
  }
}

// ---------------------------------------------------------------- EdgeConv 2

__global__ __launch_bounds__(256) void k_conv2(
    const float* __restrict__ x1, const int* __restrict__ idxb,
    const _Float16* __restrict__ Wct, const float* __restrict__ bc,
    _Float16* __restrict__ xcat) {
  __shared__ alignas(16) _Float16 wcL[128 * 128];
  __shared__ alignas(16) _Float16 eL[8 * 32 * 128];
  int tid = threadIdx.x, lane = tid & 31, wave = tid >> 5;
  // async DMA of Wc (32KB) into LDS; overlaps with the e2 gather below
  for (int i = tid; i < 128 * 128 / 8; i += 256)
    async_ld_b128(wcL + i * 8, Wct + i * 8);

  int p = blockIdx.x * 8 + wave;
  {  // build e2 rows (lane = edge row)
    int r  = lane;
    int nb = idxb[(size_t)p * KNN + (r < KNN ? r : 0)];
    const float* xi = x1 + (size_t)p * 64;
    const float* xj = x1 + (size_t)nb * 64;
    _Float16* row = eL + wave * 32 * 128 + r * 128;
#pragma unroll
    for (int c = 0; c < 64; c += 4) {
      float4 a = *(const float4*)(xi + c);
      float4 b4 = *(const float4*)(xj + c);
      row[c + 0] = (_Float16)a.x; row[c + 1] = (_Float16)a.y;
      row[c + 2] = (_Float16)a.z; row[c + 3] = (_Float16)a.w;
      row[64 + c + 0] = (_Float16)(b4.x - a.x);
      row[64 + c + 1] = (_Float16)(b4.y - a.y);
      row[64 + c + 2] = (_Float16)(b4.z - a.z);
      row[64 + c + 3] = (_Float16)(b4.w - a.w);
    }
  }
  async_wait0();
  __syncthreads();

  const _Float16* myE = eL + wave * 32 * 128;
  v8f zero = {};
  v8f c2[2][8];
#pragma unroll
  for (int m = 0; m < 2; ++m)
#pragma unroll
    for (int nt = 0; nt < 8; ++nt) c2[m][nt] = zero;

#pragma unroll
  for (int ks = 0; ks < 4; ++ks) {
    v16h a0 = load_frag(myE, 128, ks * 32, lane);
    v16h a1 = load_frag(myE + 16 * 128, 128, ks * 32, lane);
#pragma unroll
    for (int nt = 0; nt < 8; ++nt) {
      v16h bf = load_frag(wcL + nt * 16 * 128, 128, ks * 32, lane);
      c2[0][nt] = wmma16(a0, bf, c2[0][nt]);
      c2[1][nt] = wmma16(a1, bf, c2[1][nt]);
    }
  }
#pragma unroll
  for (int nt = 0; nt < 8; ++nt) {
    float m = fmaxf(tile_colmax(c2[0][nt]), tile_colmax(c2[1][nt]));
    int col = nt * 16 + (lane & 15);
    m += bc[col];
    if (lane < 16) xcat[(size_t)p * 192 + 64 + col] = (_Float16)m;
  }
}

// ------------------------------------------- 192->1024 GEMM + global max-pool

__global__ __launch_bounds__(256) void k_gemm3(const _Float16* __restrict__ xcat,
                                               const _Float16* __restrict__ Wlt,
                                               const float* __restrict__ bl,
                                               float* __restrict__ out2) {
  __shared__ alignas(16) _Float16 aL[128 * 192];      // 48KB A tile (128 points)
  __shared__ alignas(16) _Float16 wlL[8 * 16 * 192];  // 48KB Wl chunk (8 N-tiles)
  __shared__ float red[8 * 128];
  int tid = threadIdx.x, lane = tid & 31, wave = tid >> 5;
  int p0 = blockIdx.x * 128;   // 128 consecutive points, all in one graph
  int b  = p0 / NP;

  // async DMA of the 128-point A tile
  for (int i = tid; i < 128 * 192 / 8; i += 256)
    async_ld_b128(aL + i * 8, xcat + (size_t)p0 * 192 + i * 8);

  const _Float16* myA = aL + wave * 16 * 192;  // this wave's 16-point M tile
  v8f zero = {};

  for (int pass = 0; pass < 8; ++pass) {  // 8 N-tiles (128 cols) per pass
    // async DMA of this pass's Wl chunk (shared by all 8 waves)
    for (int i = tid; i < 8 * 16 * 192 / 8; i += 256)
      async_ld_b128(wlL + i * 8, Wlt + (size_t)pass * 8 * 16 * 192 + i * 8);
    async_wait0();
    __syncthreads();

    v8f c[8];
#pragma unroll
    for (int nt = 0; nt < 8; ++nt) c[nt] = zero;
#pragma unroll
    for (int ks = 0; ks < 6; ++ks) {
      v16h a = load_frag(myA, 192, ks * 32, lane);
#pragma unroll
      for (int nt = 0; nt < 8; ++nt) {
        v16h bf = load_frag(wlL + nt * 16 * 192, 192, ks * 32, lane);
        c[nt] = wmma16(a, bf, c[nt]);
      }
    }
#pragma unroll
    for (int nt = 0; nt < 8; ++nt) {
      float m = tile_colmax(c[nt]);
      if (lane < 16) red[wave * 128 + nt * 16 + (lane & 15)] = m;
    }
    __syncthreads();
    if (tid < 128) {
      float m = red[tid];
#pragma unroll
      for (int w = 1; w < 8; ++w) m = fmaxf(m, red[w * 128 + tid]);
      m += bl[pass * 128 + tid];
      gmaxf(out2 + (size_t)b * 1024 + pass * 128 + tid, m);
    }
    __syncthreads();  // wlL reused next pass
  }
}

// ---------------------------------------------------------------- MLP head

__global__ __launch_bounds__(256) void k_head(const float* __restrict__ out2,
                                              const float* __restrict__ Wa,
                                              const float* __restrict__ ba,
                                              const float* __restrict__ Wb,
                                              const float* __restrict__ bb,
                                              const float* __restrict__ Wo,
                                              const float* __restrict__ bo,
                                              float* __restrict__ out) {
  __shared__ float o2[1024], h1[512], h2s[256], lg[NCLS], stat[2];
  int b = blockIdx.x, tid = threadIdx.x;
  for (int i = tid; i < 1024; i += 256) o2[i] = out2[b * 1024 + i];
  __syncthreads();
  for (int j = tid; j < 512; j += 256) {
    float acc = ba[j];
    for (int k = 0; k < 1024; ++k) acc += o2[k] * Wa[k * 512 + j];
    h1[j] = fmaxf(acc, 0.0f);
  }
  __syncthreads();
  if (tid < 256) {
    float acc = bb[tid];
    for (int k = 0; k < 512; ++k) acc += h1[k] * Wb[k * 256 + tid];
    h2s[tid] = fmaxf(acc, 0.0f);
  }
  __syncthreads();
  if (tid < NCLS) {
    float acc = bo[tid];
    for (int k = 0; k < 256; ++k) acc += h2s[k] * Wo[k * NCLS + tid];
    lg[tid] = acc;
  }
  __syncthreads();
  if (tid == 0) {
    float mx = lg[0];
    for (int j = 1; j < NCLS; ++j) mx = fmaxf(mx, lg[j]);
    float s = 0.0f;
    for (int j = 0; j < NCLS; ++j) s += expf(lg[j] - mx);
    stat[0] = mx; stat[1] = logf(s);
  }
  __syncthreads();
  if (tid < NCLS) out[b * NCLS + tid] = lg[tid] - stat[0] - stat[1];
}

// ---------------------------------------------------------------- launcher

extern "C" void kernel_launch(void* const* d_in, const int* in_sizes, int n_in,
                              void* d_out, int out_size, void* d_ws, size_t ws_size,
                              hipStream_t stream) {
  (void)in_sizes; (void)n_in; (void)out_size; (void)ws_size;
  const float* pos  = (const float*)d_in[0];
  const float* feat = (const float*)d_in[1];
  const float* W1   = (const float*)d_in[2];
  const float* b1   = (const float*)d_in[3];
  const float* g1   = (const float*)d_in[4];
  const float* be1  = (const float*)d_in[5];
  const float* m1   = (const float*)d_in[6];
  const float* v1   = (const float*)d_in[7];
  const float* W2   = (const float*)d_in[8];
  const float* b2   = (const float*)d_in[9];
  const float* g2   = (const float*)d_in[10];
  const float* be2  = (const float*)d_in[11];
  const float* m2   = (const float*)d_in[12];
  const float* v2   = (const float*)d_in[13];
  const float* W3   = (const float*)d_in[14];
  const float* b3   = (const float*)d_in[15];
  const float* Wc   = (const float*)d_in[16];
  const float* bc   = (const float*)d_in[17];
  const float* Wl   = (const float*)d_in[18];
  const float* bl   = (const float*)d_in[19];
  const float* Wa   = (const float*)d_in[20];
  const float* ba   = (const float*)d_in[21];
  const float* Wb   = (const float*)d_in[22];
  const float* bb_  = (const float*)d_in[23];
  const float* Wo   = (const float*)d_in[24];
  const float* bo   = (const float*)d_in[25];

  char* w = (char*)d_ws;
  size_t off = 0;
  auto alloc = [&](size_t bytes) -> void* {
    void* pp = (void*)(w + off);
    off += (bytes + 255) & ~(size_t)255;
    return pp;
  };
  float*     x0   = (float*)alloc((size_t)NPTS * 4 * sizeof(float));
  float*     sq   = (float*)alloc((size_t)NPTS * sizeof(float));
  int*       idxb = (int*)alloc((size_t)NPTS * KNN * sizeof(int));
  float*     x1   = (float*)alloc((size_t)NPTS * 64 * sizeof(float));
  _Float16*  xcat = (_Float16*)alloc((size_t)NPTS * 192 * sizeof(_Float16));
  float*     out2 = (float*)alloc((size_t)BB * 1024 * sizeof(float));
  _Float16*  W1t  = (_Float16*)alloc(64 * 32 * 2);
  _Float16*  W2t  = (_Float16*)alloc(64 * 64 * 2);
  _Float16*  W3t  = (_Float16*)alloc(64 * 64 * 2);
  _Float16*  Wct  = (_Float16*)alloc(128 * 128 * 2);
  _Float16*  Wlt  = (_Float16*)alloc(1024 * 192 * 2);
  float*     b1f  = (float*)alloc(64 * 4);
  float*     b2f  = (float*)alloc(64 * 4);

  hipLaunchKernelGGL(k_prep_points, dim3(NPTS / 256), dim3(256), 0, stream,
                     pos, feat, x0, sq);
  hipLaunchKernelGGL(k_prep_weights, dim3(1), dim3(256), 0, stream,
                     W1, b1, g1, be1, m1, v1, W2, b2, g2, be2, m2, v2,
                     W3, Wc, Wl, W1t, W2t, W3t, Wct, Wlt, b1f, b2f, out2);
  hipLaunchKernelGGL(k_knn, dim3(NPTS / 256), dim3(256), 0, stream, x0, sq, idxb);
  hipLaunchKernelGGL(k_conv1, dim3(NPTS / 8), dim3(256), 0, stream,
                     x0, idxb, W1t, W2t, W3t, b1f, b2f, b3, x1, xcat);
  hipLaunchKernelGGL(k_conv2, dim3(NPTS / 8), dim3(256), 0, stream,
                     x1, idxb, Wct, bc, xcat);
  hipLaunchKernelGGL(k_gemm3, dim3(NPTS / 128), dim3(256), 0, stream,
                     xcat, Wlt, bl, out2);
  hipLaunchKernelGGL(k_head, dim3(BB), dim3(256), 0, stream,
                     out2, Wa, ba, Wb, bb_, Wo, bo, (float*)d_out);
}
